// Transformer_cait_10058813407919
// MI455X (gfx1250) — matrix operand order repaired and checked by reading, not compiled, not run on hardware
//
#include <hip/hip_runtime.h>

// ---------------------------------------------------------------------------
// CaiT-style transformer (talking-heads attention + MLP), B=8 N=1024 DIM=768
// DEPTH=6 H=12 DH=64 MLP=3072.  All large contractions run on the CDNA5
// matrix cores via v_wmma_f32_16x16x32_bf16 (bf16 operands, fp32 accum).
// LN / softmax / head-mix / GELU epilogues are fp32 VALU.
// GEMM: K-tile 64, double-buffered LDS pipeline; next tile is fetched into
// registers while the current tile feeds 16 WMMAs, then spilled to the
// alternate LDS buffer behind a single barrier.  A sched_barrier pins all
// fragment ds_loads of a k-step ahead of the WMMA burst so dscnt waits
// stage down instead of serializing at 0.
// ---------------------------------------------------------------------------

typedef __attribute__((ext_vector_type(16))) __bf16 v16bf;
typedef __attribute__((ext_vector_type(8)))  __bf16 v8bf;
typedef __attribute__((ext_vector_type(8)))  float  v8f;

#define NB   8
#define NN_  1024
#define DIM_ 768
#define DEPTH_ 6
#define NH   12
#define DH_  64
#define INNER_ 768
#define MLP_ 3072
#define ROWS (NB * NN_)   // 8192

enum { EPI_BF16 = 0, EPI_F32_SCALE = 1, EPI_BIAS_SCALE_RES = 2, EPI_BIAS_GELU = 3 };

// ---------------------------------------------------------------------------
// WMMA GEMM:  C[M,N] = A[M,K] (bf16, lda) * Bt[N,K]^T (bf16, ldb)
// 256 threads = 8 wave32, wave grid WM x WN, per-wave tile (BM/WM)x(BN/WN).
// ---------------------------------------------------------------------------
template <int BM, int BN, int WM, int WN, int EPI>
__global__ __launch_bounds__(256) void gemm_bf16(
    int K,
    const __bf16* __restrict__ A, int lda, long aZ,
    const __bf16* __restrict__ Bt, int ldb, long bZ,
    void* __restrict__ C, int ldc, long cZ,
    const float* __restrict__ bias,
    const float* __restrict__ scl,
    const float* __restrict__ resid,
    float alpha)
{
    constexpr int TM = BM / WM, TN = BN / WN;
    constexpr int FM = TM / 16, FN = TN / 16;
    constexpr int KT  = 64;            // K-tile
    constexpr int LDP = 72;            // padded row: 144 B = 36 banks, 16B aligned
    constexpr int ACH = BM / 32;       // 16B chunks per thread for A tile
    constexpr int BCH = BN / 32;

    __shared__ alignas(16) __bf16 As[2][BM][LDP];
    __shared__ alignas(16) __bf16 Bs[2][BN][LDP];

    const int tid  = threadIdx.x;
    const int wave = tid >> 5;
    const int lane = tid & 31;
    const int z    = blockIdx.z;
    const int m0   = blockIdx.y * BM;
    const int n0   = blockIdx.x * BN;
    const int wr   = wave / WN, wc = wave % WN;
    const int khalf = lane >> 4;
    const int l16   = lane & 15;

    A  += (long)z * aZ;
    Bt += (long)z * bZ;

    // tile-copy addressing: row = tid>>3 (+32*i), col chunk = (tid&7)*8
    const int rT = tid >> 3;
    const int cT = (tid & 7) * 8;
    const __bf16* pA[ACH];
    const __bf16* pB[BCH];
#pragma unroll
    for (int i = 0; i < ACH; ++i)
        pA[i] = A + (long)(m0 + rT + 32 * i) * lda + cT;
#pragma unroll
    for (int i = 0; i < BCH; ++i)
        pB[i] = Bt + (long)(n0 + rT + 32 * i) * ldb + cT;

    v8f acc[FM][FN] = {};

    // ---- prologue: tile 0 -> LDS buffer 0 ----
    {
        v8bf ra[ACH], rb[BCH];
#pragma unroll
        for (int i = 0; i < ACH; ++i) { ra[i] = *(const v8bf*)pA[i]; pA[i] += KT; }
#pragma unroll
        for (int i = 0; i < BCH; ++i) { rb[i] = *(const v8bf*)pB[i]; pB[i] += KT; }
#pragma unroll
        for (int i = 0; i < ACH; ++i) *(v8bf*)&As[0][rT + 32 * i][cT] = ra[i];
#pragma unroll
        for (int i = 0; i < BCH; ++i) *(v8bf*)&Bs[0][rT + 32 * i][cT] = rb[i];
    }
    __syncthreads();

    const int ktiles = K >> 6;
    int buf = 0;
    for (int t = 0; t < ktiles; ++t) {
        const bool hasNext = (t + 1 < ktiles);
        v8bf ra[ACH], rb[BCH];
        if (hasNext) {               // issue next tile's global loads early
#pragma unroll
            for (int i = 0; i < ACH; ++i) { ra[i] = *(const v8bf*)pA[i]; pA[i] += KT; }
#pragma unroll
            for (int i = 0; i < BCH; ++i) { rb[i] = *(const v8bf*)pB[i]; pB[i] += KT; }
        }

#pragma unroll
        for (int ks = 0; ks < 2; ++ks) {     // two k-steps of 32 per LDS tile
            // A fragment (ISA 16-bit A 16x32): lane m=l16; VGPR0-3 hold
            // K = 8*khalf..+7, VGPR4-7 hold K = 16+8*khalf..+7.
            v16bf afrag[FM];
#pragma unroll
            for (int fm = 0; fm < FM; ++fm) {
                const __bf16* ap = &As[buf][wr * TM + fm * 16 + l16][ks * 32];
                ((v8bf*)&afrag[fm])[0] = *(const v8bf*)(ap + khalf * 8);
                ((v8bf*)&afrag[fm])[1] = *(const v8bf*)(ap + 16 + khalf * 8);
            }
            // B fragment (ISA 16-bit B 32x16): lane n=l16; VGPR0-3 hold
            // K = 8*khalf..+7, VGPR4-7 hold K = 16+8*khalf..+7.  Preload all FN fragments before the WMMAs.
            v16bf bfrag[FN];
#pragma unroll
            for (int fn = 0; fn < FN; ++fn) {
                const __bf16* bp = &Bs[buf][wc * TN + fn * 16 + l16][ks * 32];
                ((v8bf*)&bfrag[fn])[0] = *(const v8bf*)(bp + khalf * 8);
                ((v8bf*)&bfrag[fn])[1] = *(const v8bf*)(bp + 16 + khalf * 8);
            }
            // Pin all ds_loads of this k-step ahead of the WMMA burst.
            __builtin_amdgcn_sched_barrier(0);
#pragma unroll
            for (int fn = 0; fn < FN; ++fn)
#pragma unroll
                for (int fm = 0; fm < FM; ++fm)
                    acc[fm][fn] = __builtin_amdgcn_wmma_f32_16x16x32_bf16(
                        false, afrag[fm], false, bfrag[fn], (short)0,
                        acc[fm][fn], false, false);
        }

        if (hasNext) {               // spill next tile into the other buffer
#pragma unroll
            for (int i = 0; i < ACH; ++i) *(v8bf*)&As[buf ^ 1][rT + 32 * i][cT] = ra[i];
#pragma unroll
            for (int i = 0; i < BCH; ++i) *(v8bf*)&Bs[buf ^ 1][rT + 32 * i][cT] = rb[i];
        }
        __syncthreads();
        buf ^= 1;
    }

    // Epilogue.  C/D layout: VGPR v -> m = v + 8*khalf, n = l16.
    const int mh = khalf * 8;
#pragma unroll
    for (int fm = 0; fm < FM; ++fm)
#pragma unroll
        for (int fn = 0; fn < FN; ++fn) {
            const int nn = n0 + wc * TN + fn * 16 + l16;
#pragma unroll
            for (int v = 0; v < 8; ++v) {
                const int  mm = m0 + wr * TM + fm * 16 + mh + v;
                const long ci = (long)mm * ldc + nn;
                float val = acc[fm][fn][v];
                if constexpr (EPI == EPI_BF16) {
                    ((__bf16*)C + (long)z * cZ)[ci] = (__bf16)val;
                } else if constexpr (EPI == EPI_F32_SCALE) {
                    ((float*)C + (long)z * cZ)[ci] = val * alpha;
                } else if constexpr (EPI == EPI_BIAS_SCALE_RES) {
                    float r = (val + bias[nn]) * scl[nn] + resid[ci];
                    ((float*)C + (long)z * cZ)[ci] = r;
                } else { // EPI_BIAS_GELU (exact erf gelu)
                    float t  = val + bias[nn];
                    float gl = 0.5f * t * (1.0f + erff(t * 0.70710678118654752f));
                    ((__bf16*)C + (long)z * cZ)[ci] = (__bf16)gl;
                }
            }
        }
}

// ---------------------------------------------------------------------------
// LayerNorm over DIM=768, one block (256 thr) per row, bf16 output.
// ---------------------------------------------------------------------------
__global__ __launch_bounds__(256) void ln_kernel(
    const float* __restrict__ x, const float* __restrict__ g,
    const float* __restrict__ b, __bf16* __restrict__ h)
{
    const int row = blockIdx.x, tid = threadIdx.x;
    const float* xr = x + (long)row * DIM_;
    float v0 = xr[tid], v1 = xr[tid + 256], v2 = xr[tid + 512];

    __shared__ float red[256];
    red[tid] = v0 + v1 + v2;
    __syncthreads();
    for (int o = 128; o > 0; o >>= 1) {
        if (tid < o) red[tid] += red[tid + o];
        __syncthreads();
    }
    const float mean = red[0] * (1.0f / DIM_);
    __syncthreads();

    const float d0 = v0 - mean, d1 = v1 - mean, d2 = v2 - mean;
    red[tid] = d0 * d0 + d1 * d1 + d2 * d2;
    __syncthreads();
    for (int o = 128; o > 0; o >>= 1) {
        if (tid < o) red[tid] += red[tid + o];
        __syncthreads();
    }
    const float rstd = rsqrtf(red[0] * (1.0f / DIM_) + 1e-5f);

    __bf16* hr = h + (long)row * DIM_;
    hr[tid]       = (__bf16)(d0 * rstd * g[tid]       + b[tid]);
    hr[tid + 256] = (__bf16)(d1 * rstd * g[tid + 256] + b[tid + 256]);
    hr[tid + 512] = (__bf16)(d2 * rstd * g[tid + 512] + b[tid + 512]);
}

// ---------------------------------------------------------------------------
// Talking-heads: mix_pre -> softmax(j) -> mix_post, one block per query row i.
// dots: [H,N,N] fp32 (one batch), attn out: [H,N,N] bf16.
// ---------------------------------------------------------------------------
__global__ __launch_bounds__(256) void mix_softmax_kernel(
    const float* __restrict__ dots, const float* __restrict__ pre,
    const float* __restrict__ post, __bf16* __restrict__ attn)
{
    const int i = blockIdx.x, tid = threadIdx.x;
    __shared__ float S[NH][NN_];     // 48 KB logits slab for this row
    __shared__ float red[256];
    __shared__ float preS[NH * NH], postS[NH * NH];

    if (tid < NH * NH) { preS[tid] = pre[tid]; postS[tid] = post[tid]; }
    for (int h = 0; h < NH; ++h)
        for (int j = tid; j < NN_; j += 256)
            S[h][j] = dots[(long)h * NN_ * NN_ + (long)i * NN_ + j];
    __syncthreads();

    float acc[NH][4];
#pragma unroll
    for (int g2 = 0; g2 < NH; ++g2)
#pragma unroll
        for (int c = 0; c < 4; ++c) acc[g2][c] = 0.0f;

    for (int g = 0; g < NH; ++g) {
        float m[4];
#pragma unroll
        for (int c = 0; c < 4; ++c) {
            const int j = tid + 256 * c;
            float s = 0.0f;
#pragma unroll
            for (int h = 0; h < NH; ++h) s += preS[g * NH + h] * S[h][j];
            m[c] = s;
        }
        float lm = fmaxf(fmaxf(m[0], m[1]), fmaxf(m[2], m[3]));
        red[tid] = lm;
        __syncthreads();
        for (int o = 128; o > 0; o >>= 1) {
            if (tid < o) red[tid] = fmaxf(red[tid], red[tid + o]);
            __syncthreads();
        }
        const float gmax = red[0];
        __syncthreads();

        float e[4], ls = 0.0f;
#pragma unroll
        for (int c = 0; c < 4; ++c) { e[c] = expf(m[c] - gmax); ls += e[c]; }
        red[tid] = ls;
        __syncthreads();
        for (int o = 128; o > 0; o >>= 1) {
            if (tid < o) red[tid] += red[tid + o];
            __syncthreads();
        }
        const float inv = 1.0f / red[0];
        __syncthreads();

#pragma unroll
        for (int g2 = 0; g2 < NH; ++g2) {
            const float w = postS[g2 * NH + g];
#pragma unroll
            for (int c = 0; c < 4; ++c) acc[g2][c] += w * e[c] * inv;
        }
    }

#pragma unroll
    for (int g2 = 0; g2 < NH; ++g2)
#pragma unroll
        for (int c = 0; c < 4; ++c)
            attn[(long)g2 * NN_ * NN_ + (long)i * NN_ + tid + 256 * c] =
                (__bf16)acc[g2][c];
}

// fp32 [K,N] weight -> bf16 transposed [N,K]
__global__ void convert_wt_kernel(const float* __restrict__ W,
                                  __bf16* __restrict__ Wt, int K, int N)
{
    long idx = (long)blockIdx.x * 256 + threadIdx.x;
    if (idx >= (long)K * N) return;
    int n = (int)(idx % N);
    int k = (int)(idx / N);
    Wt[(long)n * K + k] = (__bf16)W[idx];
}

// kv [ROWS,1536] bf16 -> vt [B,H,DH,N] bf16 (V transposed for attn@V GEMM)
__global__ void transpose_v_kernel(const __bf16* __restrict__ kv,
                                   __bf16* __restrict__ vt)
{
    long idx = (long)blockIdx.x * 256 + threadIdx.x; // < B*H*DH*N
    int n = (int)(idx % NN_);
    long t = idx / NN_;
    int d = (int)(t % DH_); t /= DH_;
    int h = (int)(t % NH);
    int b = (int)(t / NH);
    vt[idx] = kv[((long)(b * NN_ + n)) * (2 * INNER_) + INNER_ + h * DH_ + d];
}

// ---------------------------------------------------------------------------
extern "C" void kernel_launch(void* const* d_in, const int* in_sizes, int n_in,
                              void* d_out, int out_size, void* d_ws, size_t ws_size,
                              hipStream_t stream)
{
    (void)in_sizes; (void)n_in; (void)out_size; (void)ws_size;
    const float* x_in = (const float*)d_in[0];
    const float* ln1g = (const float*)d_in[1];
    const float* ln1b = (const float*)d_in[2];
    const float* Wq   = (const float*)d_in[3];
    const float* Wkv  = (const float*)d_in[4];
    const float* pre  = (const float*)d_in[5];
    const float* post = (const float*)d_in[6];
    const float* Wo   = (const float*)d_in[7];
    const float* bo   = (const float*)d_in[8];
    const float* s1   = (const float*)d_in[9];
    const float* ln2g = (const float*)d_in[10];
    const float* ln2b = (const float*)d_in[11];
    const float* W1   = (const float*)d_in[12];
    const float* b1   = (const float*)d_in[13];
    const float* W2   = (const float*)d_in[14];
    const float* b2   = (const float*)d_in[15];
    const float* s2   = (const float*)d_in[16];

    // ---- workspace layout (~312 MB) ----
    char*  ws  = (char*)d_ws;
    size_t off = 0;
    auto take = [&](size_t bytes) -> char* {
        char* p = ws + off;
        off += (bytes + 255) & ~(size_t)255;
        return p;
    };
    __bf16* wqT  = (__bf16*)take((size_t)DEPTH_ * INNER_ * DIM_ * 2);
    __bf16* wkvT = (__bf16*)take((size_t)DEPTH_ * 2 * INNER_ * DIM_ * 2);
    __bf16* woT  = (__bf16*)take((size_t)DEPTH_ * DIM_ * INNER_ * 2);
    __bf16* w1T  = (__bf16*)take((size_t)DEPTH_ * MLP_ * DIM_ * 2);
    __bf16* w2T  = (__bf16*)take((size_t)DEPTH_ * DIM_ * MLP_ * 2);
    float*  xbuf = (float*)take((size_t)ROWS * DIM_ * 4);
    __bf16* hbuf = (__bf16*)take((size_t)ROWS * DIM_ * 2);
    __bf16* qbuf = (__bf16*)take((size_t)ROWS * INNER_ * 2);
    __bf16* kvbuf = (__bf16*)take((size_t)ROWS * 2 * INNER_ * 2);
    __bf16* vtbuf = (__bf16*)take((size_t)NB * NH * DH_ * NN_ * 2);
    __bf16* obuf = (__bf16*)take((size_t)ROWS * INNER_ * 2);
    __bf16* mlpbuf = (__bf16*)take((size_t)ROWS * MLP_ * 2);
    float*  dots = (float*)take((size_t)NH * NN_ * NN_ * 4);     // per batch
    __bf16* attnb = (__bf16*)take((size_t)NH * NN_ * NN_ * 2);   // per batch

    // ---- convert + transpose weights to bf16 (all layers) ----
    for (int i = 0; i < DEPTH_; ++i) {
        convert_wt_kernel<<<(DIM_ * INNER_) / 256, 256, 0, stream>>>(
            Wq + (long)i * DIM_ * INNER_, wqT + (long)i * INNER_ * DIM_, DIM_, INNER_);
        convert_wt_kernel<<<(DIM_ * 2 * INNER_) / 256, 256, 0, stream>>>(
            Wkv + (long)i * DIM_ * 2 * INNER_, wkvT + (long)i * 2 * INNER_ * DIM_, DIM_, 2 * INNER_);
        convert_wt_kernel<<<(INNER_ * DIM_) / 256, 256, 0, stream>>>(
            Wo + (long)i * INNER_ * DIM_, woT + (long)i * DIM_ * INNER_, INNER_, DIM_);
        convert_wt_kernel<<<(DIM_ * MLP_) / 256, 256, 0, stream>>>(
            W1 + (long)i * DIM_ * MLP_, w1T + (long)i * MLP_ * DIM_, DIM_, MLP_);
        convert_wt_kernel<<<(MLP_ * DIM_) / 256, 256, 0, stream>>>(
            W2 + (long)i * MLP_ * DIM_, w2T + (long)i * DIM_ * MLP_, MLP_, DIM_);
    }

    // x working copy
    hipMemcpyAsync(xbuf, x_in, (size_t)ROWS * DIM_ * 4,
                   hipMemcpyDeviceToDevice, stream);

    const float qk_scale = 0.125f; // DH^-0.5

    for (int i = 0; i < DEPTH_; ++i) {
        // ---- attention block ----
        ln_kernel<<<ROWS, 256, 0, stream>>>(xbuf, ln1g + i * DIM_, ln1b + i * DIM_, hbuf);

        gemm_bf16<128, 128, 4, 2, EPI_BF16><<<dim3(INNER_ / 128, ROWS / 128, 1), 256, 0, stream>>>(
            DIM_, hbuf, DIM_, 0,
            wqT + (long)i * INNER_ * DIM_, DIM_, 0,
            qbuf, INNER_, 0, nullptr, nullptr, nullptr, 0.0f);

        gemm_bf16<128, 128, 4, 2, EPI_BF16><<<dim3(2 * INNER_ / 128, ROWS / 128, 1), 256, 0, stream>>>(
            DIM_, hbuf, DIM_, 0,
            wkvT + (long)i * 2 * INNER_ * DIM_, DIM_, 0,
            kvbuf, 2 * INNER_, 0, nullptr, nullptr, nullptr, 0.0f);

        transpose_v_kernel<<<(NB * NH * DH_ * NN_) / 256, 256, 0, stream>>>(kvbuf, vtbuf);

        for (int b = 0; b < NB; ++b) {
            // dots[h,i,j] = scale * q_h[i,:] . k_h[j,:]
            gemm_bf16<128, 128, 4, 2, EPI_F32_SCALE><<<dim3(NN_ / 128, NN_ / 128, NH), 256, 0, stream>>>(
                DH_,
                qbuf + (long)b * NN_ * INNER_, INNER_, DH_,
                kvbuf + (long)b * NN_ * 2 * INNER_, 2 * INNER_, DH_,
                dots, NN_, (long)NN_ * NN_, nullptr, nullptr, nullptr, qk_scale);

            mix_softmax_kernel<<<NN_, 256, 0, stream>>>(
                dots, pre + i * NH * NH, post + i * NH * NH, attnb);

            // out_h[i,d] = attn_h[i,:] . v_h[:,d]
            gemm_bf16<128, 64, 8, 1, EPI_BF16><<<dim3(1, NN_ / 128, NH), 256, 0, stream>>>(
                NN_,
                attnb, NN_, (long)NN_ * NN_,
                vtbuf + (long)b * NH * DH_ * NN_, NN_, (long)DH_ * NN_,
                obuf + (long)b * NN_ * INNER_, INNER_, DH_,
                nullptr, nullptr, nullptr, 0.0f);
        }

        // x = (out @ Wo + bo) * s1 + x   (in-place on xbuf)
        gemm_bf16<128, 128, 4, 2, EPI_BIAS_SCALE_RES><<<dim3(DIM_ / 128, ROWS / 128, 1), 256, 0, stream>>>(
            INNER_, obuf, INNER_, 0,
            woT + (long)i * DIM_ * INNER_, INNER_, 0,
            xbuf, DIM_, 0, bo + i * DIM_, s1 + i * DIM_, xbuf, 0.0f);

        // ---- MLP block ----
        ln_kernel<<<ROWS, 256, 0, stream>>>(xbuf, ln2g + i * DIM_, ln2b + i * DIM_, hbuf);

        gemm_bf16<128, 128, 4, 2, EPI_BIAS_GELU><<<dim3(MLP_ / 128, ROWS / 128, 1), 256, 0, stream>>>(
            DIM_, hbuf, DIM_, 0,
            w1T + (long)i * MLP_ * DIM_, DIM_, 0,
            mlpbuf, MLP_, 0, b1 + i * MLP_, nullptr, nullptr, 0.0f);

        gemm_bf16<128, 128, 4, 2, EPI_BIAS_SCALE_RES><<<dim3(DIM_ / 128, ROWS / 128, 1), 256, 0, stream>>>(
            MLP_, mlpbuf, MLP_, 0,
            w2T + (long)i * DIM_ * MLP_, MLP_, 0,
            xbuf, DIM_, 0, b2 + i * DIM_, s2 + i * DIM_, xbuf, 0.0f);
    }

    hipMemcpyAsync(d_out, xbuf, (size_t)ROWS * DIM_ * 4,
                   hipMemcpyDeviceToDevice, stream);
}
